// DynamicConv2d_10067403342463
// MI455X (gfx1250) — compile-verified, matrix operand order checked by reading, not run on hardware
//
#include <hip/hip_runtime.h>

#define BATCH 16
#define CIN   128
#define HH    128
#define WW    128
#define COUTC 128
#define NEXP  4
#define ACH   32
#define TEMPF 30.0f
#define KTOT  (9 * 128)   // 1152
#define ASTR  40          // padded LDS row stride (halfwords) -> 80B
#define BSTR  40
#define NSTEPS 36         // 9 taps * 4 channel chunks of 32

typedef __attribute__((ext_vector_type(16))) __bf16 v16bf;
typedef __attribute__((ext_vector_type(8)))  float  v8f;
typedef __attribute__((ext_vector_type(4)))  int    v4i;

// pointer-to-vector types in explicit address spaces for the async builtin
typedef __attribute__((address_space(1))) v4i* gas_v4i_p;   // global
typedef __attribute__((address_space(3))) v4i* las_v4i_p;   // LDS

#if defined(__has_builtin)
#  if __has_builtin(__builtin_amdgcn_global_load_async_to_lds_b128)
#    define HAS_ASYNC_LDS 1
#  else
#    define HAS_ASYNC_LDS 0
#  endif
#  if __has_builtin(__builtin_amdgcn_s_wait_asynccnt)
#    define HAS_WAIT_ASYNC_BUILTIN 1
#  else
#    define HAS_WAIT_ASYNC_BUILTIN 0
#  endif
#else
#  define HAS_ASYNC_LDS 0
#  define HAS_WAIT_ASYNC_BUILTIN 0
#endif

__device__ __forceinline__ void wait_async_lds() {
#if HAS_ASYNC_LDS
#  if HAS_WAIT_ASYNC_BUILTIN
    __builtin_amdgcn_s_wait_asynccnt(0);
#  else
    asm volatile("s_wait_asynccnt 0x0" ::: "memory");
#  endif
#endif
}

__device__ __forceinline__ unsigned short f2bf(float f) {
    union { float f; unsigned u; } c; c.f = f;
    unsigned u = c.u;
    unsigned r = u + 0x7FFFu + ((u >> 16) & 1u);   // round-to-nearest-even
    return (unsigned short)(r >> 16);
}

// ---------------- Kernel 1: global average pool (B,C) ----------------
__global__ __launch_bounds__(256) void pool_kernel(const float* __restrict__ x,
                                                   float* __restrict__ pooled) {
    __shared__ float red[256];
    int bc = blockIdx.x;                       // b*CIN + c
    const float4* p = (const float4*)(x + (size_t)bc * (HH * WW));
    int t = threadIdx.x;
    float s = 0.f;
#pragma unroll
    for (int i = 0; i < 16; ++i) {
        float4 v = p[t + i * 256];
        s += v.x + v.y + v.z + v.w;
    }
    red[t] = s;
    __syncthreads();
    for (int off = 128; off > 0; off >>= 1) {
        if (t < off) red[t] += red[t + off];
        __syncthreads();
    }
    if (t == 0) pooled[bc] = red[0] * (1.0f / (HH * WW));
}

// ------------- Kernel 2: attention MLP + softmax + agg bias -------------
__global__ __launch_bounds__(256) void attn_kernel(const float* __restrict__ pooled,
                                                   const float* __restrict__ aw1,
                                                   const float* __restrict__ ab1,
                                                   const float* __restrict__ aw2,
                                                   const float* __restrict__ ab2,
                                                   const float* __restrict__ bias,
                                                   float* __restrict__ attn_out,
                                                   float* __restrict__ aggb_out) {
    __shared__ float hbuf[BATCH * ACH];
    __shared__ float attn_s[BATCH * NEXP];
    int t = threadIdx.x;

    for (int i = t; i < BATCH * ACH; i += 256) {
        int b = i / ACH, a = i % ACH;
        float acc = ab1[a];
        const float* pr = pooled + b * CIN;
        const float* wr = aw1 + a * CIN;
        for (int c = 0; c < CIN; ++c) acc += pr[c] * wr[c];
        hbuf[i] = acc > 0.f ? acc : 0.f;
    }
    __syncthreads();

    if (t < BATCH * NEXP) {
        int b = t / NEXP, k = t % NEXP;
        float acc = ab2[k];
        for (int a = 0; a < ACH; ++a) acc += hbuf[b * ACH + a] * aw2[k * ACH + a];
        attn_s[t] = acc * (1.0f / TEMPF);
    }
    __syncthreads();

    if (t < BATCH) {
        float m = attn_s[t * NEXP];
        for (int k = 1; k < NEXP; ++k) m = fmaxf(m, attn_s[t * NEXP + k]);
        float e[NEXP]; float sum = 0.f;
        for (int k = 0; k < NEXP; ++k) { e[k] = __expf(attn_s[t * NEXP + k] - m); sum += e[k]; }
        float inv = 1.0f / sum;
        for (int k = 0; k < NEXP; ++k) {
            attn_s[t * NEXP + k] = e[k] * inv;
            attn_out[t * NEXP + k] = e[k] * inv;
        }
    }
    __syncthreads();

    for (int i = t; i < BATCH * COUTC; i += 256) {
        int b = i / COUTC, o = i % COUTC;
        float acc = 0.f;
        for (int k = 0; k < NEXP; ++k) acc += attn_s[b * NEXP + k] * bias[k * COUTC + o];
        aggb_out[i] = acc;
    }
}

// ----- Kernel 3: per-sample weight aggregation -> bf16, layout [b][cout][tap*128+c] -----
__global__ __launch_bounds__(256) void aggw_kernel(const float* __restrict__ weight,
                                                   const float* __restrict__ attn,
                                                   unsigned short* __restrict__ aggw) {
    int i = blockIdx.x * 256 + threadIdx.x;     // total = 16*128*9*128
    int c   = i & 127;
    int tap = (i >> 7) % 9;
    int o   = (i / KTOT) & 127;
    int b   = i / (KTOT * COUTC);
    const size_t ESTRIDE = (size_t)COUTC * CIN * 9;
    const float* wb = weight + ((size_t)o * CIN + c) * 9 + tap;
    const float* ab = attn + b * NEXP;
    float acc = ab[0] * wb[0] + ab[1] * wb[ESTRIDE] + ab[2] * wb[2 * ESTRIDE] + ab[3] * wb[3 * ESTRIDE];
    aggw[((size_t)(b * COUTC + o)) * KTOT + tap * CIN + c] = f2bf(acc);
}

// ------------- Kernel 4: implicit-GEMM conv via bf16 WMMA -------------
// Block: 256 thr (8 wave32). Tile M=128 (cout) x N=128 (full W row).
// K = 1152 in 36 steps of 32 (tap-major, 32 channels per step).
// Double-buffered LDS; A-chunk staged via async global->LDS when available.

__device__ __forceinline__ void stage_a(const unsigned short* __restrict__ aggw_b,
                                        int tap, int cc, unsigned short* lbuf, int t) {
#pragma unroll
    for (int r = 0; r < 2; ++r) {
        int idx  = t + r * 256;          // 0..511
        int cout = idx >> 2;
        int seg  = (idx & 3) * 8;        // 8-halfword segment within the 32-K chunk
        const unsigned short* g = aggw_b + (size_t)cout * KTOT + tap * CIN + cc + seg;
        unsigned short* l = lbuf + cout * ASTR + seg;
#if HAS_ASYNC_LDS
        __builtin_amdgcn_global_load_async_to_lds_b128(
            (gas_v4i_p)g, (las_v4i_p)l, 0, 0);
#else
        *(uint4*)l = *(const uint4*)g;
#endif
    }
}

__device__ __forceinline__ void load_b_regs(const float* __restrict__ xb, int hi, bool rowok,
                                            int dw, int kc, int nb, int cc, float* regs) {
    const float* xrow = xb + ((size_t)(cc + kc) * HH + hi) * WW;
#pragma unroll
    for (int j = 0; j < 16; ++j) {
        int wsrc = nb + j + dw;
        float v = 0.f;
        if (rowok && wsrc >= 0 && wsrc < WW) v = xrow[wsrc];
        regs[j] = v;
    }
}

__device__ __forceinline__ void store_b_lds(unsigned short* lbuf, int kc, int nb,
                                            const float* regs) {
#pragma unroll
    for (int j = 0; j < 16; ++j)
        lbuf[(nb + j) * BSTR + kc] = f2bf(regs[j]);   // lanes span 32 banks: conflict-free
}

__global__ __launch_bounds__(256) void conv_wmma_kernel(const float* __restrict__ x,
                                                        const unsigned short* __restrict__ aggw,
                                                        const float* __restrict__ aggb,
                                                        float* __restrict__ out) {
    __shared__ alignas(16) unsigned short lds_a[2][COUTC * ASTR];   // 2 x 10 KB
    __shared__ alignas(16) unsigned short lds_b[2][WW * BSTR];      // 2 x 10 KB
    __shared__ float aggb_s[COUTC];

    int gb = blockIdx.x;
    int h  = gb & 127;
    int b  = gb >> 7;

    int t    = threadIdx.x;
    int lane = t & 31;
    int wave = t >> 5;
    int wm = (wave & 3) * 32;     // cout base of this wave (M tile 32)
    int wn = (wave >> 2) * 64;    // n base of this wave    (N tile 64)

    const float* xb = x + (size_t)b * CIN * HH * WW;
    const unsigned short* aggw_b = aggw + (size_t)b * COUTC * KTOT;

    v8f acc[2][4] = {};           // 2x4 of 16x16 f32 tiles

    // B staging mapping: kc = lane (channel within 32-chunk), 16 consecutive n per thread
    int kc = t & 31;
    int nb = (t >> 5) * 16;
    float bregs[16];

    // fragment addressing per documented 16-bit 16x32 layout
    int lrow = lane & 15;
    int koff = (lane & 16) ? 8 : 0;

    if (t < COUTC) aggb_s[t] = aggb[b * COUTC + t];

    // ---- prologue: stage step 0 into buffer 0 ----
    stage_a(aggw_b, 0, 0, lds_a[0], t);
    load_b_regs(xb, h - 1, (h - 1) >= 0, -1, kc, nb, 0, bregs);
    store_b_lds(lds_b[0], kc, nb, bregs);
    wait_async_lds();
    __syncthreads();

    int buf = 0;
    for (int s = 0; s < NSTEPS; ++s) {
        int nxt = s + 1;
        bool more = (nxt < NSTEPS);
        if (more) {
            int tap = nxt >> 2, ccs = (nxt & 3) * 32;
            int dh = tap / 3 - 1, dw = tap % 3 - 1;
            int hi = h + dh;
            load_b_regs(xb, hi, (hi >= 0) && (hi < HH), dw, kc, nb, ccs, bregs);
            stage_a(aggw_b, tap, ccs, lds_a[buf ^ 1], t);
        }

        // ---- compute: 2 A-frags + 4 B-frags -> 8 WMMAs ----
        {
            union Frag { uint4 u[2]; v16bf v; };
            Frag af[2], bfr[4];
            const unsigned short* la = lds_a[buf];
            const unsigned short* lb = lds_b[buf];
#pragma unroll
            for (int i = 0; i < 2; ++i) {
                int row = wm + i * 16 + lrow;
                af[i].u[0] = *(const uint4*)&la[row * ASTR + koff];
                af[i].u[1] = *(const uint4*)&la[row * ASTR + koff + 16];
            }
#pragma unroll
            for (int j = 0; j < 4; ++j) {
                int col = wn + j * 16 + lrow;
                bfr[j].u[0] = *(const uint4*)&lb[col * BSTR + koff];
                bfr[j].u[1] = *(const uint4*)&lb[col * BSTR + koff + 16];
            }
#pragma unroll
            for (int i = 0; i < 2; ++i)
#pragma unroll
                for (int j = 0; j < 4; ++j)
                    acc[i][j] = __builtin_amdgcn_wmma_f32_16x16x32_bf16(
                        false, af[i].v, false, bfr[j].v, (short)0, acc[i][j], false, false);
        }

        if (more) store_b_lds(lds_b[buf ^ 1], kc, nb, bregs);
        wait_async_lds();
        __syncthreads();
        buf ^= 1;
    }

    // ---- epilogue: + bias, non-temporal stores (out is write-once stream) ----
    int mlanebase = (lane >> 4) * 8;   // lanes 16..31 hold M+8
#pragma unroll
    for (int i = 0; i < 2; ++i) {
#pragma unroll
        for (int j = 0; j < 4; ++j) {
            int n = wn + j * 16 + lrow;
#pragma unroll
            for (int r = 0; r < 8; ++r) {
                int m = wm + i * 16 + mlanebase + r;
                float val = acc[i][j][r] + aggb_s[m];
                __builtin_nontemporal_store(val, &out[(((size_t)(b * COUTC + m) * HH + h) * WW) + n]);
            }
        }
    }
}

// ---------------------------------------------------------------------
extern "C" void kernel_launch(void* const* d_in, const int* in_sizes, int n_in,
                              void* d_out, int out_size, void* d_ws, size_t ws_size,
                              hipStream_t stream) {
    const float* x    = (const float*)d_in[0];
    const float* wgt  = (const float*)d_in[1];
    const float* bias = (const float*)d_in[2];
    const float* aw1  = (const float*)d_in[3];
    const float* ab1  = (const float*)d_in[4];
    const float* aw2  = (const float*)d_in[5];
    const float* ab2  = (const float*)d_in[6];
    float* out = (float*)d_out;

    char* ws = (char*)d_ws;
    float* attn   = (float*)ws;                          // 64 f
    float* pooled = (float*)(ws + 256);                  // 2048 f
    float* aggb   = (float*)(ws + 256 + 8192);           // 2048 f
    unsigned short* aggw = (unsigned short*)(ws + 256 + 8192 + 8192);  // 16*128*1152 bf16 (~4.5MB)

    pool_kernel<<<BATCH * CIN, 256, 0, stream>>>(x, pooled);
    attn_kernel<<<1, 256, 0, stream>>>(pooled, aw1, ab1, aw2, ab2, bias, attn, aggb);
    aggw_kernel<<<(BATCH * COUTC * 9 * CIN) / 256, 256, 0, stream>>>(wgt, attn, aggw);
    conv_wmma_kernel<<<BATCH * HH, 256, 0, stream>>>(x, aggw, aggb, out);
}
